// StructureModuleTE_15221364097680
// MI455X (gfx1250) — compile-verified, matrix operand order checked by reading, not run on hardware
//
#include <hip/hip_runtime.h>
#include <hip/hip_bf16.h>
#include <math.h>

#define B_    2
#define N_    512
#define CS_   384
#define CZ_   128
#define H_    12
#define HD_   32
#define L_    8
#define CRES_ 128

// output element offsets (float32)
#define OA 0
#define OU 14336
#define OB 28672
#define OG 45056
#define OP 176128
#define OS 219136

typedef __bf16 bf16_t;
typedef __bf16 v8bf  __attribute__((ext_vector_type(8)));
typedef __bf16 v16bf __attribute__((ext_vector_type(16)));
typedef float  v8f   __attribute__((ext_vector_type(8)));

__device__ __forceinline__ v16bf ld_frag(const bf16_t* p, int g){
  v8bf lo = *(const v8bf*)(p + g*8);
  v8bf hi = *(const v8bf*)(p + 16 + g*8);
  return __builtin_shufflevector(lo, hi, 0,1,2,3,4,5,6,7,8,9,10,11,12,13,14,15);
}
__device__ __forceinline__ v8f zf8(){
  v8f z = {0.f,0.f,0.f,0.f,0.f,0.f,0.f,0.f};
  return z;
}
__device__ __forceinline__ v8f wmma_bf16(v16bf a, v16bf b, v8f c){
  return __builtin_amdgcn_wmma_f32_16x16x32_bf16(false, a, false, b, (short)0, c, false, false);
}

// ---------------- generic GEMM: out[m][n] = act(A[m][:]·W[n][:] + bias[n] + resid) -------------
// One wave computes a 16x64 tile (A-fragment reused across 4 WMMAs). Out-of-range weight
// rows are CLAMPED (not masked) so the inner loop has no EXEC divergence; only the
// epilogue guards n < Nc.
__device__ __forceinline__ void gemm_epi(v8f acc, int n, int Nc, int m0, int g,
    const float* __restrict__ bias, const float* __restrict__ resid,
    float* __restrict__ outF, bf16_t* __restrict__ outB, int act)
{
  if (n >= Nc) return;
  float bn = bias ? bias[n] : 0.f;
  #pragma unroll
  for (int r2 = 0; r2 < 8; r2++){
    int m = m0 + 8*g + r2;
    float v = acc[r2] + bn;
    if (resid) v += resid[(size_t)m*Nc + n];
    if (act == 1) v = fmaxf(v, 0.f);
    else if (act == 2) v = 1.f/(1.f + __expf(-v));
    if (outF) outF[(size_t)m*Nc + n] = v;
    if (outB) outB[(size_t)m*Nc + n] = (bf16_t)v;
  }
}

__global__ __launch_bounds__(256) void gemm_kernel(
    const bf16_t* __restrict__ A, const bf16_t* __restrict__ W,
    const float* __restrict__ bias, const float* __restrict__ resid,
    float* __restrict__ outF, bf16_t* __restrict__ outB,
    int M, int Nc, int K, int act, int tn4)
{
  int wid = blockIdx.x*8 + (threadIdx.x >> 5);
  int tm = M >> 4;
  if (wid >= tm*tn4) return;
  int lane = threadIdx.x & 31;
  int g = lane >> 4, l15 = lane & 15;
  int tile_n = wid % tn4, tile_m = wid / tn4;
  int m0 = tile_m << 4, n0 = tile_n << 6;
  const bf16_t* ap = A + (size_t)(m0 + l15)*K;
  const bf16_t* wp0; const bf16_t* wp1; const bf16_t* wp2; const bf16_t* wp3;
  {
    int r0 = n0 + l15;        if (r0 > Nc-1) r0 = Nc-1;
    int r1 = n0 + 16 + l15;   if (r1 > Nc-1) r1 = Nc-1;
    int r2 = n0 + 32 + l15;   if (r2 > Nc-1) r2 = Nc-1;
    int r3 = n0 + 48 + l15;   if (r3 > Nc-1) r3 = Nc-1;
    wp0 = W + (size_t)r0*K; wp1 = W + (size_t)r1*K;
    wp2 = W + (size_t)r2*K; wp3 = W + (size_t)r3*K;
  }
  v8f a0 = zf8(), a1 = zf8(), a2 = zf8(), a3 = zf8();
  for (int k = 0; k < K; k += 32){
    v16bf af = ld_frag(ap + k, g);
    a0 = wmma_bf16(af, ld_frag(wp0 + k, g), a0);
    a1 = wmma_bf16(af, ld_frag(wp1 + k, g), a1);
    a2 = wmma_bf16(af, ld_frag(wp2 + k, g), a2);
    a3 = wmma_bf16(af, ld_frag(wp3 + k, g), a3);
  }
  gemm_epi(a0, n0 +      l15, Nc, m0, g, bias, resid, outF, outB, act);
  gemm_epi(a1, n0 + 16 + l15, Nc, m0, g, bias, resid, outF, outB, act);
  gemm_epi(a2, n0 + 32 + l15, Nc, m0, g, bias, resid, outF, outB, act);
  gemm_epi(a3, n0 + 48 + l15, Nc, m0, g, bias, resid, outF, outB, act);
}

// ---------------- bias-matrix GEMM: one wave computes a full 16x96 row-block ----------------
// (A-fragment reused across all 6 channel tiles), scattered to (b, layer, head, q, k) bf16.
__global__ __launch_bounds__(256) void bmat_kernel(
    const bf16_t* __restrict__ zn, const bf16_t* __restrict__ wb,
    const float* __restrict__ bb, bf16_t* __restrict__ bm)
{
  const int tm = (B_*N_*N_) >> 4;   // 32768 row-blocks
  int wid = blockIdx.x*8 + (threadIdx.x >> 5);
  if (wid >= tm) return;
  int lane = threadIdx.x & 31;
  int g = lane >> 4, l15 = lane & 15;
  int m0 = wid << 4;
  const bf16_t* ap = zn + (size_t)(m0 + l15)*CZ_;
  v8f acc[6];
  #pragma unroll
  for (int t = 0; t < 6; t++) acc[t] = zf8();
  #pragma unroll
  for (int k = 0; k < CZ_; k += 32){
    v16bf af = ld_frag(ap + k, g);
    #pragma unroll
    for (int t = 0; t < 6; t++){
      v16bf wf = ld_frag(wb + (size_t)(t*16 + l15)*CZ_ + k, g);
      acc[t] = wmma_bf16(af, wf, acc[t]);
    }
  }
  int bidx = m0 >> 18;                 // / (N*N)
  int rem  = m0 & (N_*N_ - 1);
  int q = rem >> 9;
  int kr0 = rem & (N_-1);
  #pragma unroll
  for (int t = 0; t < 6; t++){
    int n = t*16 + l15;
    int ii = n / 12, hh = n % 12;
    float bn = bb[n];
    bf16_t* dst = bm + ((size_t)(bidx*96 + ii*12 + hh))*((size_t)N_*N_) + (size_t)q*N_;
    #pragma unroll
    for (int r2 = 0; r2 < 8; r2++)
      dst[kr0 + 8*g + r2] = (bf16_t)(acc[t][r2] + bn);
  }
}

// ---------------- transposed flash attention: one wave per (b,h, 16 queries) ----------------
__global__ __launch_bounds__(256) void attn_kernel(
    const bf16_t* __restrict__ qb, const bf16_t* __restrict__ kb,
    const bf16_t* __restrict__ vt, const bf16_t* __restrict__ bm,
    const float* __restrict__ mb, float* __restrict__ y, int layer)
{
  int wid = blockIdx.x*8 + (threadIdx.x >> 5);
  if (wid >= B_*H_*(N_/16)) return;
  int lane = threadIdx.x & 31;
  int g = lane >> 4, l15 = lane & 15;
  int qt = wid & 31;
  int bh = wid >> 5;
  int h = bh % H_, b = bh / H_;
  int q0 = qt << 4;
  const bf16_t* qp = qb + (size_t)bh*N_*HD_;
  const bf16_t* kp = kb + (size_t)bh*N_*HD_;
  const bf16_t* vp = vt + (size_t)bh*HD_*N_;
  int qrow = q0 + l15;
  const bf16_t* brow = bm + ((size_t)((b*L_ + layer)*H_ + h)*N_ + qrow)*N_;
  const float* mrow = mb + b*N_;
  v16bf qf = ld_frag(qp + (size_t)qrow*HD_, g);   // B-operand: lane = query column
  float m_run = -1e30f, l_run = 0.f;
  v8f o0 = zf8(), o1 = zf8();                     // O^T tiles: channels 0-15 / 16-31
  for (int k0 = 0; k0 < N_; k0 += 32){
    v16bf ak0 = ld_frag(kp + (size_t)(k0 + l15)*HD_, g);       // keys k0..k0+15
    v16bf ak1 = ld_frag(kp + (size_t)(k0 + 16 + l15)*HD_, g);  // keys k0+16..k0+31
    v8f s0 = wmma_bf16(ak0, qf, zf8());   // S^T[key][q]
    v8f s1 = wmma_bf16(ak1, qf, zf8());
    v8bf b0 = *(const v8bf*)(brow + k0 + g*8);
    v8bf b1 = *(const v8bf*)(brow + k0 + 16 + g*8);
    const float* m0p = mrow + k0 + g*8;
    float tmax = -1e30f;
    #pragma unroll
    for (int r2 = 0; r2 < 8; r2++){
      s0[r2] += (float)b0[r2] + m0p[r2];
      s1[r2] += (float)b1[r2] + m0p[16 + r2];
      tmax = fmaxf(tmax, fmaxf(s0[r2], s1[r2]));
    }
    tmax = fmaxf(tmax, __shfl_xor(tmax, 16));
    float m_new = fmaxf(m_run, tmax);
    float alpha = __expf(m_run - m_new);
    float lsum = 0.f;
    v16bf pf;     // P as B-operand: halves 0..7 = tile0 regs, 8..15 = tile1 regs (free!)
    #pragma unroll
    for (int r2 = 0; r2 < 8; r2++){
      float p0 = __expf(s0[r2] - m_new);
      float p1 = __expf(s1[r2] - m_new);
      lsum += p0 + p1;
      pf[r2]     = (bf16_t)p0;
      pf[8 + r2] = (bf16_t)p1;
    }
    lsum += __shfl_xor(lsum, 16);
    l_run = l_run*alpha + lsum;
    m_run = m_new;
    v16bf av0 = ld_frag(vp + (size_t)l15*N_ + k0, g);        // V^T rows c=0..15
    v16bf av1 = ld_frag(vp + (size_t)(16 + l15)*N_ + k0, g); // c=16..31
    #pragma unroll
    for (int r2 = 0; r2 < 8; r2++){ o0[r2] *= alpha; o1[r2] *= alpha; }
    o0 = wmma_bf16(av0, pf, o0);
    o1 = wmma_bf16(av1, pf, o1);
  }
  float inv = 1.f / fmaxf(l_run, 1e-30f);
  float* yr = y + (size_t)(b*N_ + qrow)*CS_ + h*HD_;
  #pragma unroll
  for (int r2 = 0; r2 < 8; r2++){
    yr[8*g + r2]      = o0[r2]*inv;
    yr[16 + 8*g + r2] = o1[r2]*inv;
  }
}

// ---------------- layernorm: one wave per row ----------------
__global__ void ln_kernel(const float* __restrict__ x, const float* __restrict__ gm,
                          const float* __restrict__ bt, float* __restrict__ outF,
                          bf16_t* __restrict__ outB, int rows, int C)
{
  int wid = blockIdx.x*(blockDim.x >> 5) + (threadIdx.x >> 5);
  if (wid >= rows) return;
  int lane = threadIdx.x & 31;
  const float* xr = x + (size_t)wid*C;
  float s = 0.f, ss = 0.f;
  for (int c = lane; c < C; c += 32){ float v = xr[c]; s += v; ss += v*v; }
  for (int o = 16; o > 0; o >>= 1){ s += __shfl_xor(s, o); ss += __shfl_xor(ss, o); }
  float mean = s / C;
  float var = ss / C - mean*mean;
  float rs = rsqrtf(var + 1e-5f);
  for (int c = lane; c < C; c += 32){
    float v = (xr[c] - mean)*rs*gm[c] + bt[c];
    if (outF) outF[(size_t)wid*C + c] = v;
    if (outB) outB[(size_t)wid*C + c] = (bf16_t)v;
  }
}

// ---------------- small elementwise kernels ----------------
__global__ void f2b_kernel(const float* __restrict__ s, bf16_t* __restrict__ d, int n){
  int i = blockIdx.x*blockDim.x + threadIdx.x;
  if (i < n) d[i] = (bf16_t)s[i];
}
__global__ void relu_f2b_kernel(const float* __restrict__ s, bf16_t* __restrict__ d, int n){
  int i = blockIdx.x*blockDim.x + threadIdx.x;
  if (i < n) d[i] = (bf16_t)fmaxf(s[i], 0.f);
}
__global__ void gy_kernel(const float* __restrict__ g, const float* __restrict__ y,
                          bf16_t* __restrict__ d, int n){
  int i = blockIdx.x*blockDim.x + threadIdx.x;
  if (i < n) d[i] = (bf16_t)(g[i]*y[i]);
}
__global__ void maskbias_kernel(const int* __restrict__ m, float* __restrict__ d, int n){
  int i = blockIdx.x*blockDim.x + threadIdx.x;
  if (i < n) d[i] = (m[i] == 0) ? -1e9f : 0.f;
}
__global__ void repack_kernel(const float* __restrict__ qkv, bf16_t* __restrict__ q,
                              bf16_t* __restrict__ k, bf16_t* __restrict__ v){
  int idx = blockIdx.x*blockDim.x + threadIdx.x;
  if (idx >= B_*H_*N_) return;
  int n = idx % N_;
  int bh = idx / N_;
  int h = bh % H_;
  const float* src = qkv + ((size_t)((bh/H_)*N_ + n))*1152 + h*96;
  const float sc = 0.17677669529663687f;   // 1/sqrt(32)
  bf16_t* qd = q + (size_t)idx*HD_;
  bf16_t* kd = k + (size_t)idx*HD_;
  for (int c = 0; c < HD_; c++){
    qd[c] = (bf16_t)(src[c]*sc);
    kd[c] = (bf16_t)(src[HD_ + c]);
    v[((size_t)bh*HD_ + c)*N_ + n] = (bf16_t)(src[2*HD_ + c]);
  }
}
__global__ void bb_kernel(const float* __restrict__ s, const float* __restrict__ w,
                          const float* __restrict__ b, float* __restrict__ o){
  int row = blockIdx.x;
  int j = threadIdx.x;
  if (j >= 9) return;
  const float* sr = s + (size_t)row*CS_;
  const float* wr = w + (size_t)j*CS_;
  float acc = b[j];
  for (int c = 0; c < CS_; c++) acc += sr[c]*wr[c];
  o[row*9 + j] = acc;
}

// ---------------- per-residue geometry (rigid frames + FK + atoms) ----------------
__device__ __forceinline__ void mm3(const float* A, const float* Bm, float* C){
  #pragma unroll
  for (int i = 0; i < 3; i++)
    #pragma unroll
    for (int j = 0; j < 3; j++)
      C[i*3+j] = A[i*3]*Bm[j] + A[i*3+1]*Bm[3+j] + A[i*3+2]*Bm[6+j];
}
__device__ __forceinline__ void mv3(const float* A, const float* v, float* o){
  #pragma unroll
  for (int i = 0; i < 3; i++)
    o[i] = A[i*3]*v[0] + A[i*3+1]*v[1] + A[i*3+2]*v[2];
}

__global__ void geom_kernel(const float* __restrict__ ang14, const float* __restrict__ bb9,
    const int* __restrict__ aatype, const float* __restrict__ deff,
    const int* __restrict__ gidx, const float* __restrict__ amask,
    const float* __restrict__ lit, float* __restrict__ out)
{
  int r = blockIdx.x*blockDim.x + threadIdx.x;
  if (r >= B_*N_) return;
  float sin_a[8], cos_a[8];
  sin_a[0] = 0.f; cos_a[0] = 1.f;
  #pragma unroll
  for (int j = 0; j < 7; j++){
    float a0 = ang14[r*14 + 2*j];
    float a1 = ang14[r*14 + 2*j + 1];
    out[OU + r*14 + 2*j]     = a0;
    out[OU + r*14 + 2*j + 1] = a1;
    float dn = sqrtf(fmaxf(a0*a0 + a1*a1, 1e-8f));
    float sj = a0/dn, cj = a1/dn;
    out[OA + r*14 + 2*j]     = sj;
    out[OA + r*14 + 2*j + 1] = cj;
    sin_a[j+1] = sj; cos_a[j+1] = cj;
  }
  float nv[3], cav[3], cvv[3];
  #pragma unroll
  for (int i = 0; i < 3; i++){ nv[i]=bb9[r*9+i]; cav[i]=bb9[r*9+3+i]; cvv[i]=bb9[r*9+6+i]; }
  float nx[3] = {nv[0]-cav[0], nv[1]-cav[1], nv[2]-cav[2]};
  float cx = cvv[0]-cav[0], cy = cvv[1]-cav[1], cz = cvv[2]-cav[2];
  float nrm = sqrtf(1e-7f + cx*cx + cy*cy);
  float s1 = -cy/nrm, c1 = cx/nrm;
  float R1[9] = {c1,-s1,0.f, s1,c1,0.f, 0.f,0.f,1.f};
  float nrm2 = sqrtf(1e-7f + cx*cx + cy*cy + cz*cz);
  float s2 = cz/nrm2, c2 = sqrtf(cx*cx + cy*cy)/nrm2;
  float R2[9] = {c2,0.f,s2, 0.f,1.f,0.f, -s2,0.f,c2};
  float Rc[9]; mm3(R2, R1, Rc);
  float n2[3]; mv3(Rc, nx, n2);
  float ny = n2[1], nzv = n2[2];
  float nrm3 = sqrtf(1e-7f + ny*ny + nzv*nzv);
  float sn = -nzv/nrm3, cn = ny/nrm3;
  float Rn[9] = {1.f,0.f,0.f, 0.f,cn,-sn, 0.f,sn,cn};
  float Rt[9]; mm3(Rn, Rc, Rt);
  float Rbb[9];
  #pragma unroll
  for (int i = 0; i < 3; i++)
    #pragma unroll
    for (int j = 0; j < 3; j++) Rbb[i*3+j] = Rt[j*3+i];
  float tbb[3] = {cav[0]*10.f, cav[1]*10.f, cav[2]*10.f};
  {
    float* ob = out + OB + (size_t)r*16;
    for (int i = 0; i < 3; i++){
      ob[i*4]=Rbb[i*3]; ob[i*4+1]=Rbb[i*3+1]; ob[i*4+2]=Rbb[i*3+2]; ob[i*4+3]=tbb[i];
    }
    ob[12]=0.f; ob[13]=0.f; ob[14]=0.f; ob[15]=1.f;
  }
  int aa = aatype[r];
  float Rall[8][9], tall[8][3];
  for (int g2 = 0; g2 < 8; g2++){
    const float* d4 = deff + ((size_t)aa*8 + g2)*16;
    float dR[9], dt[3];
    for (int i = 0; i < 3; i++){
      dR[i*3]=d4[i*4]; dR[i*3+1]=d4[i*4+1]; dR[i*3+2]=d4[i*4+2]; dt[i]=d4[i*4+3];
    }
    float sa = sin_a[g2], ca = cos_a[g2];
    float Ra[9] = {1.f,0.f,0.f, 0.f,ca,-sa, 0.f,sa,ca};
    mm3(dR, Ra, Rall[g2]);
    tall[g2][0]=dt[0]; tall[g2][1]=dt[1]; tall[g2][2]=dt[2];
  }
  float R5[9], t5[3], R6[9], t6[3], R7[9], t7[3];
  mm3(Rall[4], Rall[5], R5); mv3(Rall[4], tall[5], t5);
  t5[0]+=tall[4][0]; t5[1]+=tall[4][1]; t5[2]+=tall[4][2];
  mm3(R5, Rall[6], R6); mv3(R5, tall[6], t6);
  t6[0]+=t5[0]; t6[1]+=t5[1]; t6[2]+=t5[2];
  mm3(R6, Rall[7], R7); mv3(R6, tall[7], t7);
  t7[0]+=t6[0]; t7[1]+=t6[1]; t7[2]+=t6[2];
  for (int i = 0; i < 9; i++){ Rall[5][i]=R5[i]; Rall[6][i]=R6[i]; Rall[7][i]=R7[i]; }
  for (int i = 0; i < 3; i++){ tall[5][i]=t5[i]; tall[6][i]=t6[i]; tall[7][i]=t7[i]; }
  float Rg[8][9], tg[8][3];
  for (int g2 = 0; g2 < 8; g2++){
    mm3(Rbb, Rall[g2], Rg[g2]);
    mv3(Rbb, tall[g2], tg[g2]);
    tg[g2][0]+=tbb[0]; tg[g2][1]+=tbb[1]; tg[g2][2]+=tbb[2];
    float* og = out + OG + ((size_t)r*8 + g2)*16;
    for (int i = 0; i < 3; i++){
      og[i*4]=Rg[g2][i*3]; og[i*4+1]=Rg[g2][i*3+1]; og[i*4+2]=Rg[g2][i*3+2]; og[i*4+3]=tg[g2][i];
    }
    og[12]=0.f; og[13]=0.f; og[14]=0.f; og[15]=1.f;
  }
  for (int at = 0; at < 14; at++){
    int g2 = gidx[aa*14 + at];
    const float* lp = lit + ((size_t)aa*14 + at)*3;
    float mk = amask[aa*14 + at];
    for (int i = 0; i < 3; i++){
      float p = Rg[g2][i*3]*lp[0] + Rg[g2][i*3+1]*lp[1] + Rg[g2][i*3+2]*lp[2] + tg[g2][i];
      out[OP + ((size_t)r*14 + at)*3 + i] = p*mk;
    }
  }
}

// ======================================================================================
extern "C" void kernel_launch(void* const* d_in, const int* in_sizes, int n_in,
                              void* d_out, int out_size, void* d_ws, size_t ws_size,
                              hipStream_t stream)
{
  (void)in_sizes; (void)n_in; (void)out_size; (void)ws_size;
  const float* s_in   = (const float*)d_in[0];
  const float* z_in   = (const float*)d_in[1];
  const int*   aatype = (const int*)d_in[2];
  const int*   maskp  = (const int*)d_in[3];
  const float* ln_s_g = (const float*)d_in[4];
  const float* ln_s_b = (const float*)d_in[5];
  const float* ln_z_g = (const float*)d_in[6];
  const float* ln_z_b = (const float*)d_in[7];
  const float* w_in   = (const float*)d_in[8];
  const float* b_in   = (const float*)d_in[9];
  const float* w_b    = (const float*)d_in[10];
  const float* b_b    = (const float*)d_in[11];
  const float* attn_ln_g = (const float*)d_in[12];
  const float* attn_ln_b = (const float*)d_in[13];
  const float* attn_wqkv = (const float*)d_in[14];
  const float* attn_wo = (const float*)d_in[15];
  const float* attn_bo = (const float*)d_in[16];
  const float* attn_wg = (const float*)d_in[17];
  const float* attn_bg = (const float*)d_in[18];
  const float* tr_ln_g = (const float*)d_in[19];
  const float* tr_ln_b = (const float*)d_in[20];
  const float* tr_w1 = (const float*)d_in[21];
  const float* tr_b1 = (const float*)d_in[22];
  const float* tr_w2 = (const float*)d_in[23];
  const float* tr_b2 = (const float*)d_in[24];
  const float* w_bb = (const float*)d_in[25];
  const float* b_bb = (const float*)d_in[26];
  const float* ar_win = (const float*)d_in[27];
  const float* ar_bin = (const float*)d_in[28];
  const float* ar_winit = (const float*)d_in[29];
  const float* ar_binit = (const float*)d_in[30];
  const float* ar_w1 = (const float*)d_in[31];
  const float* ar_b1 = (const float*)d_in[32];
  const float* ar_w2 = (const float*)d_in[33];
  const float* ar_b2 = (const float*)d_in[34];
  const float* ar_wout = (const float*)d_in[35];
  const float* ar_bout = (const float*)d_in[36];
  const float* deff = (const float*)d_in[37];
  const int*   gidx = (const int*)d_in[38];
  const float* amask = (const float*)d_in[39];
  const float* litp = (const float*)d_in[40];
  float* out = (float*)d_out;

  char* wsp = (char*)d_ws;
  auto alloc = [&](size_t elems, size_t esz)->void*{
    void* p = (void*)wsp;
    wsp += (elems*esz + 255) & ~(size_t)255;
    return p;
  };
  bf16_t* w_in16    = (bf16_t*)alloc(384*384, 2);
  bf16_t* wqkv16    = (bf16_t*)alloc((size_t)L_*1152*384, 2);
  bf16_t* wo16      = (bf16_t*)alloc((size_t)L_*384*384, 2);
  bf16_t* wg16      = (bf16_t*)alloc((size_t)L_*384*384, 2);
  bf16_t* tr1_16    = (bf16_t*)alloc((size_t)L_*384*384, 2);
  bf16_t* tr2_16    = (bf16_t*)alloc((size_t)L_*384*384, 2);
  bf16_t* wb16      = (bf16_t*)alloc(96*128, 2);
  bf16_t* arwin16   = (bf16_t*)alloc(128*384, 2);
  bf16_t* arwinit16 = (bf16_t*)alloc(128*384, 2);
  bf16_t* arw1_16   = (bf16_t*)alloc(2*128*128, 2);
  bf16_t* arw2_16   = (bf16_t*)alloc(2*128*128, 2);
  bf16_t* arwout16  = (bf16_t*)alloc(14*128, 2);
  bf16_t* zn16   = (bf16_t*)alloc((size_t)B_*N_*N_*CZ_, 2);
  bf16_t* bm16   = (bf16_t*)alloc((size_t)B_*L_*H_*N_*N_, 2);
  float*  s_init = (float*)alloc((size_t)B_*N_*CS_, 4);
  float*  s_cur  = (float*)alloc((size_t)B_*N_*CS_, 4);
  bf16_t* xn16   = (bf16_t*)alloc((size_t)B_*N_*CS_, 2);
  float*  qkv32  = (float*)alloc((size_t)B_*N_*1152, 4);
  bf16_t* qb16   = (bf16_t*)alloc((size_t)B_*H_*N_*HD_, 2);
  bf16_t* kb16   = (bf16_t*)alloc((size_t)B_*H_*N_*HD_, 2);
  bf16_t* vt16   = (bf16_t*)alloc((size_t)B_*H_*N_*HD_, 2);
  float*  y32    = (float*)alloc((size_t)B_*N_*CS_, 4);
  float*  gate32 = (float*)alloc((size_t)B_*N_*CS_, 4);
  bf16_t* gy16   = (bf16_t*)alloc((size_t)B_*N_*CS_, 2);
  bf16_t* h16    = (bf16_t*)alloc((size_t)B_*N_*CS_, 2);
  bf16_t* reluA16 = (bf16_t*)alloc((size_t)B_*N_*CS_, 2);
  bf16_t* reluB16 = (bf16_t*)alloc((size_t)B_*N_*CS_, 2);
  bf16_t* relu128 = (bf16_t*)alloc((size_t)B_*N_*CRES_, 2);
  bf16_t* rh16   = (bf16_t*)alloc((size_t)B_*N_*CRES_, 2);
  float*  a0_32  = (float*)alloc((size_t)B_*N_*CRES_, 4);
  float*  a_32   = (float*)alloc((size_t)B_*N_*CRES_, 4);
  float*  ang32  = (float*)alloc((size_t)B_*N_*14, 4);
  float*  mb32   = (float*)alloc((size_t)B_*N_, 4);
  float*  bb9_32 = (float*)alloc((size_t)B_*N_*9, 4);

  auto f2b = [&](const float* src, bf16_t* dst, size_t n){
    f2b_kernel<<<(unsigned)((n + 255)/256), 256, 0, stream>>>(src, dst, (int)n);
  };
  auto gemm = [&](const bf16_t* A, const bf16_t* Wm, const float* bias, const float* resid,
                  float* oF, bf16_t* oB, int M, int Nc, int K, int act){
    int tn4 = (Nc + 63) >> 6;
    int waves = (M >> 4)*tn4;
    gemm_kernel<<<(waves + 7) >> 3, 256, 0, stream>>>(A, Wm, bias, resid, oF, oB, M, Nc, K, act, tn4);
  };
  auto ln = [&](const float* x, const float* gmv, const float* btv,
                float* oF, bf16_t* oB, int rows, int C){
    ln_kernel<<<(rows + 7) >> 3, 256, 0, stream>>>(x, gmv, btv, oF, oB, rows, C);
  };

  // ---- weight conversion to bf16 ----
  f2b(w_in, w_in16, 384*384);
  f2b(attn_wqkv, wqkv16, (size_t)L_*1152*384);
  f2b(attn_wo, wo16, (size_t)L_*384*384);
  f2b(attn_wg, wg16, (size_t)L_*384*384);
  f2b(tr_w1, tr1_16, (size_t)L_*384*384);
  f2b(tr_w2, tr2_16, (size_t)L_*384*384);
  f2b(w_b, wb16, 96*128);
  f2b(ar_win, arwin16, 128*384);
  f2b(ar_winit, arwinit16, 128*384);
  f2b(ar_w1, arw1_16, 2*128*128);
  f2b(ar_w2, arw2_16, 2*128*128);
  f2b(ar_wout, arwout16, 14*128);

  maskbias_kernel<<<4, 256, 0, stream>>>(maskp, mb32, B_*N_);

  // ---- pair bias: LN(z) -> bf16, then WMMA GEMM scattered to (b,layer,head,q,k) ----
  ln(z_in, ln_z_g, ln_z_b, nullptr, zn16, B_*N_*N_, CZ_);
  {
    int waves = (B_*N_*N_) >> 4;    // one wave per 16x96 row-block
    bmat_kernel<<<(waves + 7) >> 3, 256, 0, stream>>>(zn16, wb16, b_b, bm16);
  }

  // ---- s path ----
  ln(s_in, ln_s_g, ln_s_b, s_init, xn16, B_*N_, CS_);
  gemm(xn16, w_in16, b_in, nullptr, s_cur, nullptr, B_*N_, 384, 384, 0);

  for (int i = 0; i < L_; i++){
    ln(s_cur, attn_ln_g + i*CS_, attn_ln_b + i*CS_, nullptr, xn16, B_*N_, CS_);
    gemm(xn16, wqkv16 + (size_t)i*1152*384, nullptr, nullptr, qkv32, nullptr, B_*N_, 1152, 384, 0);
    repack_kernel<<<96, 256, 0, stream>>>(qkv32, qb16, kb16, vt16);
    attn_kernel<<<96, 256, 0, stream>>>(qb16, kb16, vt16, bm16, mb32, y32, i);
    gemm(xn16, wg16 + (size_t)i*384*384, attn_bg + i*CS_, nullptr, gate32, nullptr, B_*N_, 384, 384, 2);
    gy_kernel<<<1536, 256, 0, stream>>>(gate32, y32, gy16, B_*N_*CS_);
    gemm(gy16, wo16 + (size_t)i*384*384, attn_bo + i*CS_, s_cur, s_cur, nullptr, B_*N_, 384, 384, 0);
    ln(s_cur, tr_ln_g + i*CS_, tr_ln_b + i*CS_, nullptr, xn16, B_*N_, CS_);
    gemm(xn16, tr1_16 + (size_t)i*384*384, tr_b1 + i*CS_, nullptr, nullptr, h16, B_*N_, 384, 384, 1);
    gemm(h16, tr2_16 + (size_t)i*384*384, tr_b2 + i*CS_, s_cur, s_cur, nullptr, B_*N_, 384, 384, 0);
  }

  // ---- angle resnet ----
  relu_f2b_kernel<<<1536, 256, 0, stream>>>(s_init, reluA16, B_*N_*CS_);
  relu_f2b_kernel<<<1536, 256, 0, stream>>>(s_cur, reluB16, B_*N_*CS_);
  gemm(reluA16, arwinit16, ar_binit, nullptr, a0_32, nullptr, B_*N_, 128, 384, 0);
  gemm(reluB16, arwin16, ar_bin, a0_32, a_32, nullptr, B_*N_, 128, 384, 0);
  for (int j = 0; j < 2; j++){
    relu_f2b_kernel<<<512, 256, 0, stream>>>(a_32, relu128, B_*N_*CRES_);
    gemm(relu128, arw1_16 + j*128*128, ar_b1 + j*128, nullptr, nullptr, rh16, B_*N_, 128, 128, 1);
    gemm(rh16, arw2_16 + j*128*128, ar_b2 + j*128, a_32, a_32, nullptr, B_*N_, 128, 128, 0);
  }
  relu_f2b_kernel<<<512, 256, 0, stream>>>(a_32, relu128, B_*N_*CRES_);
  gemm(relu128, arwout16, ar_bout, nullptr, ang32, nullptr, B_*N_, 14, 128, 0);

  // ---- backbone + geometry ----
  bb_kernel<<<B_*N_, 16, 0, stream>>>(s_cur, w_bb, b_bb, bb9_32);
  geom_kernel<<<(B_*N_ + 63)/64, 64, 0, stream>>>(ang32, bb9_32, aatype, deff, gidx, amask, litp, out);

  hipMemcpyAsync(out + OS, s_cur, (size_t)B_*N_*CS_*sizeof(float),
                 hipMemcpyDeviceToDevice, stream);
}
